// Quantize_13056700579871
// MI455X (gfx1250) — matrix operand order repaired and checked by reading, not compiled, not run on hardware
//
#include <hip/hip_runtime.h>
#include <hip/hip_bf16.h>

typedef unsigned short ushort_t;
typedef unsigned int   uint_t;

typedef __attribute__((ext_vector_type(16))) __bf16 v16bf;
typedef __attribute__((ext_vector_type(8)))  float  v8f;

#define CODE_DIM 256
#define NUM_EMB  1024
#define HW       1024          // H*W
#define NPIX     32768         // B*H*W
#define PIX_PER_BLK 32
#define A_STRIDE 264           // padded LDS row stride (ushorts), 16B-aligned rows

union FragU {
    v16bf v;
    uint4 q[2];
};

__device__ __forceinline__ ushort_t f32_to_bf16_rne(float f) {
    uint_t u = __builtin_bit_cast(uint_t, f);
    uint_t r = u + 0x7FFFu + ((u >> 16) & 1u);
    return (ushort_t)(r >> 16);
}
__device__ __forceinline__ float bf16_to_f32(ushort_t h) {
    uint_t u = ((uint_t)h) << 16;
    return __builtin_bit_cast(float, u);
}

// ---------------------------------------------------------------------------
// Prep: embeddings f32 -> bf16 hi/lo split + ||e||^2
// ---------------------------------------------------------------------------
__global__ void vq_prep_kernel(const float* __restrict__ emb,
                               ushort_t* __restrict__ e_hi,
                               ushort_t* __restrict__ e_lo,
                               float* __restrict__ enorm) {
    __shared__ float red[256];
    const int row = blockIdx.x;
    const int c   = threadIdx.x;

    float x = emb[row * CODE_DIM + c];
    ushort_t h = f32_to_bf16_rne(x);
    float hf = bf16_to_f32(h);
    ushort_t l = f32_to_bf16_rne(x - hf);
    e_hi[row * CODE_DIM + c] = h;
    e_lo[row * CODE_DIM + c] = l;

    red[c] = x * x;
    __syncthreads();
    for (int s = 128; s > 0; s >>= 1) {
        if (c < s) red[c] += red[c + s];
        __syncthreads();
    }
    if (c == 0) enorm[row] = red[0];
}

// ---------------------------------------------------------------------------
// Main: 1024 blocks x 256 threads (8 wave32 waves). Block owns 32 pixels.
// bf16-split WMMA (hi*hi + hi*lo + lo*hi ~ fp32 dot) against all 1024 codes.
// A-hi fragments kept register-resident; A-lo reloaded from LDS per j-block
// (opaque-zero offset blocks LICM so nothing spills to scratch).
// ---------------------------------------------------------------------------
__global__ void __launch_bounds__(256, 1)
vq_main_kernel(const float* __restrict__ z,
               const float* __restrict__ emb,
               const ushort_t* __restrict__ e_hi,
               const ushort_t* __restrict__ e_lo,
               const float* __restrict__ enorm,
               float* __restrict__ out_q,
               float* __restrict__ out_st,
               float* __restrict__ out_idx) {
    __shared__ float    zf32[CODE_DIM * PIX_PER_BLK];      // [c][p]  32KB
    __shared__ ushort_t ahi[PIX_PER_BLK * A_STRIDE];       // [p][k]  bf16 hi
    __shared__ ushort_t alo[PIX_PER_BLK * A_STRIDE];       // [p][k]  bf16 lo
    __shared__ float    xnorm[PIX_PER_BLK];
    __shared__ float    bestD_s[PIX_PER_BLK * 4];
    __shared__ int      bestI_s[PIX_PER_BLK * 4];
    __shared__ int      chosen[PIX_PER_BLK];

    const int t    = threadIdx.x;
    const int w    = t >> 5;          // wave id 0..7
    const int lane = t & 31;
    const int n0   = blockIdx.x * PIX_PER_BLK;
    const int bimg = n0 >> 10;        // n0 / HW
    const int hw0  = n0 & (HW - 1);

    if (t < PIX_PER_BLK) xnorm[t] = 0.0f;
    __syncthreads();

    // ---- Stage z: coalesced (lane = pixel), wave w covers c = w + 8k ----
    float part = 0.0f;
    #pragma unroll 4
    for (int kk = 0; kk < 32; ++kk) {
        int c = w + (kk << 3);
        float x = z[(size_t)(bimg * CODE_DIM + c) * HW + hw0 + lane];
        zf32[c * PIX_PER_BLK + lane] = x;
        ushort_t h = f32_to_bf16_rne(x);
        ushort_t l = f32_to_bf16_rne(x - bf16_to_f32(h));
        ahi[lane * A_STRIDE + c] = h;
        alo[lane * A_STRIDE + c] = l;
        part += x * x;
    }
    atomicAdd(&xnorm[lane], part);   // ds_add_f32
    __syncthreads();

    // ---- Per-slot pixel norms (C/D layout: M = v + 8*(lane>>4)) ----
    const int piBase = (w >> 2) * 16;
    const int half   = lane >> 4;    // 0 or 1
    float xnl[8];
    #pragma unroll
    for (int v = 0; v < 8; ++v)
        xnl[v] = xnorm[piBase + v + 8 * half];

    float bestD[8];
    int   bestI[8];
    #pragma unroll
    for (int v = 0; v < 8; ++v) { bestD[v] = __builtin_inff(); bestI[v] = 0; }

    const int arow = lane & 15;                 // A-matrix row for this lane
    const ushort_t* aHiRow = &ahi[(piBase + arow) * A_STRIDE];
    const ushort_t* aLoRow = &alo[(piBase + arow) * A_STRIDE];

    // ---- Preload ONLY the A-hi fragments (8 x 8 VGPR = 64 VGPRs resident) ----
    FragU aHf[CODE_DIM / 32];
    #pragma unroll
    for (int kc = 0; kc < CODE_DIM / 32; ++kc) {
        const int aoff = kc * 32 + half * 8;
        aHf[kc].q[0] = *(const uint4*)(aHiRow + aoff);
        aHf[kc].q[1] = *(const uint4*)(aHiRow + aoff + 16);
    }

    // ---- Sweep all embeddings in 64-wide j-blocks ----
    #pragma unroll 1
    for (int jb = 0; jb < NUM_EMB / 64; ++jb) {
        const int j0   = jb * 64 + (w & 3) * 16;
        const int jcol = j0 + (lane & 15);

        if (jb + 1 < NUM_EMB / 64) {    // global_prefetch_b8 next tile
            __builtin_prefetch(&e_hi[(size_t)(jcol + 64) * CODE_DIM], 0, 1);
            __builtin_prefetch(&e_lo[(size_t)(jcol + 64) * CODE_DIM], 0, 1);
        }

        // Opaque zero: compiler cannot prove loop-invariance of A-lo LDS
        // addresses below, so the ds_load_b128s stay here (no spill/reload).
        int zofs;
        asm volatile("s_mov_b32 %0, 0" : "=s"(zofs));

        v8f acc = {};
        #pragma unroll
        for (int kc = 0; kc < CODE_DIM / 32; ++kc) {
            const int kBase = kc * 32;
            const int aoff  = kBase + half * 8 + zofs;
            FragU aL;
            aL.q[0] = *(const uint4*)(aLoRow + aoff);
            aL.q[1] = *(const uint4*)(aLoRow + aoff + 16);

            // B fragment: column N = lane&15 (one embedding row), K = 8*half+[0..7] and 16+8*half+[0..7]
            FragU bH, bL;
            const size_t boff = (size_t)jcol * CODE_DIM + kBase + half * 8;
            bH.q[0] = *(const uint4*)(e_hi + boff);
            bH.q[1] = *(const uint4*)(e_hi + boff + 16);
            bL.q[0] = *(const uint4*)(e_lo + boff);
            bL.q[1] = *(const uint4*)(e_lo + boff + 16);

            acc = __builtin_amdgcn_wmma_f32_16x16x32_bf16(
                false, aHf[kc].v, false, bH.v, (short)0, acc, false, false);
            acc = __builtin_amdgcn_wmma_f32_16x16x32_bf16(
                false, aHf[kc].v, false, bL.v, (short)0, acc, false, false);
            acc = __builtin_amdgcn_wmma_f32_16x16x32_bf16(
                false, aL.v, false, bH.v, (short)0, acc, false, false);
        }

        const float en = enorm[jcol];
        #pragma unroll
        for (int v = 0; v < 8; ++v) {
            float d = xnl[v] - 2.0f * acc[v] + en;
            if (d < bestD[v]) { bestD[v] = d; bestI[v] = jcol; }
        }
    }

    // ---- Reduce argmin across the 16 N-lanes of each half (xor stays in half) ----
    #pragma unroll
    for (int v = 0; v < 8; ++v) {
        float bd = bestD[v];
        int   bi = bestI[v];
        #pragma unroll
        for (int m = 1; m < 16; m <<= 1) {
            float od = __shfl_xor(bd, m, 32);
            int   oi = __shfl_xor(bi, m, 32);
            if (od < bd || (od == bd && oi < bi)) { bd = od; bi = oi; }
        }
        if ((lane & 15) == 0) {
            int pb = piBase + v + 8 * half;
            bestD_s[pb * 4 + (w & 3)] = bd;
            bestI_s[pb * 4 + (w & 3)] = bi;
        }
    }
    __syncthreads();

    // ---- Final reduce across the 4 j-sub-tile waves; emit idx output ----
    if (t < PIX_PER_BLK) {
        float bd = bestD_s[t * 4];
        int   bi = bestI_s[t * 4];
        #pragma unroll
        for (int k = 1; k < 4; ++k) {
            float od = bestD_s[t * 4 + k];
            int   oi = bestI_s[t * 4 + k];
            if (od < bd || (od == bd && oi < bi)) { bd = od; bi = oi; }
        }
        chosen[t] = bi;
        out_idx[n0 + t] = (float)bi;   // idx output (int reference, float buffer)
    }
    __syncthreads();

    // ---- Gather codebook rows; write quantized + straight-through ----
    const int e = chosen[lane];
    #pragma unroll 4
    for (int kk = 0; kk < 32; ++kk) {
        int c = w + (kk << 3);
        float q  = emb[(size_t)e * CODE_DIM + c];
        float zv = zf32[c * PIX_PER_BLK + lane];
        size_t o = (size_t)(bimg * CODE_DIM + c) * HW + hw0 + lane;
        out_q[o]  = q;
        out_st[o] = (q - zv) + zv;     // matches reference fp arithmetic
    }
}

// ---------------------------------------------------------------------------
extern "C" void kernel_launch(void* const* d_in, const int* in_sizes, int n_in,
                              void* d_out, int out_size, void* d_ws, size_t ws_size,
                              hipStream_t stream) {
    (void)in_sizes; (void)n_in; (void)out_size; (void)ws_size;

    const float* z   = (const float*)d_in[0];   // (32, 256, 32, 32)
    const float* emb = (const float*)d_in[1];   // (1024, 256)

    char* ws = (char*)d_ws;
    ushort_t* e_hi  = (ushort_t*)ws;                             // 512 KB
    ushort_t* e_lo  = (ushort_t*)(ws + (size_t)NUM_EMB * CODE_DIM * 2);
    float*    enorm = (float*)(ws + (size_t)NUM_EMB * CODE_DIM * 4);

    float* out   = (float*)d_out;
    float* out_q = out;                                          // 8388608
    float* out_s = out + (size_t)NPIX * CODE_DIM;                // 8388608
    float* out_i = out + (size_t)2 * NPIX * CODE_DIM;            // 32768

    vq_prep_kernel<<<NUM_EMB, 256, 0, stream>>>(emb, e_hi, e_lo, enorm);
    vq_main_kernel<<<NPIX / PIX_PER_BLK, 256, 0, stream>>>(
        z, emb, e_hi, e_lo, enorm, out_q, out_s, out_i);
}
